// Linear_39608188404308
// MI455X (gfx1250) — matrix operand order repaired and checked by reading, not compiled, not run on hardware
//
#include <hip/hip_runtime.h>
#include <stdint.h>

// ---- CDNA5 wave32 WMMA types ----
typedef __attribute__((ext_vector_type(16))) _Float16 v16h;
typedef __attribute__((ext_vector_type(4)))  _Float16 v4h;
typedef __attribute__((ext_vector_type(8)))  float    v8f;

struct alignas(32) FragBits { uint4 lo, hi; };

#define BM 128
#define BN 128
#define BK 32
#define LDSS 40   // LDS row stride in halves (32 data + 8 pad) -> 80B rows, conflict-free b128 frag reads

__global__ __launch_bounds__(256)
void int8_linear_wmma_f16(const float* __restrict__ X,      // [M,K] fp32
                          const int8_t* __restrict__ W,     // [N,K] int8 (row-major)
                          const float* __restrict__ Sc,     // [N] per-out-channel scale
                          float* __restrict__ Out,          // [M,N] fp32
                          int M, int N, int K)
{
    // ping-pong LDS buffers: 2 * (10240 + 10240) = 40 KB
    __shared__ _Float16 As[2][BM * LDSS];
    __shared__ _Float16 Bs[2][BN * LDSS];

    const int tid  = threadIdx.x;
    const int lane = tid & 31;
    const int wave = tid >> 5;          // 8 waves
    const int wm   = wave >> 1;         // 0..3 : 32-row strip in M
    const int wn   = wave & 1;          // 0..1 : 64-col strip in N

    const int block_m = blockIdx.y * BM;
    const int block_n = blockIdx.x * BN;

    // staging indexing: each thread loads 4 elems x 4 rows per tile
    const int q  = tid & 7;             // k-quad: k = q*4
    const int r0 = tid >> 3;            // row 0..31, stepped by 32

    // fragment indexing (ISA 7.12.2 layouts, wave32)
    const int lh   = lane & 15;
    const int asel = (lane < 16) ? 0 : 8;    // A: K chunks {0..7,16..23} vs {8..15,24..31}
    const int bsel = (lane < 16) ? 0 : 8;    // B: K chunks {0..7,16..23} vs {8..15,24..31}

    v8f acc[2][4] = {};                 // 2 (M) x 4 (N) 16x16 f32 accumulators

    const float*  xbase = X + (size_t)(block_m + r0) * K + q * 4;
    const int8_t* wbase = W + (size_t)(block_n + r0) * K + q * 4;

    float4  fa[4];
    int32_t wb[4];

    // ---- prologue: stage tile 0 into buffer 0 ----
    #pragma unroll
    for (int p = 0; p < 4; ++p) {
        fa[p] = *(const float4*)(xbase + (size_t)p * 32 * K);
        wb[p] = *(const int32_t*)(wbase + (size_t)p * 32 * K);
    }
    #pragma unroll
    for (int p = 0; p < 4; ++p) {
        const int row = r0 + p * 32;
        v4h ha;
        ha.x = (_Float16)fa[p].x; ha.y = (_Float16)fa[p].y;
        ha.z = (_Float16)fa[p].z; ha.w = (_Float16)fa[p].w;
        *(v4h*)(&As[0][row * LDSS + q * 4]) = ha;
        v4h hb;
        hb.x = (_Float16)(int8_t)( wb[p]        & 0xff);
        hb.y = (_Float16)(int8_t)((wb[p] >>  8) & 0xff);
        hb.z = (_Float16)(int8_t)((wb[p] >> 16) & 0xff);
        hb.w = (_Float16)(int8_t)((wb[p] >> 24) & 0xff);
        *(v4h*)(&Bs[0][row * LDSS + q * 4]) = hb;
    }
    __syncthreads();

    const int nT = K / BK;              // 128 K-steps

    #pragma unroll 2
    for (int t = 0; t < nT; ++t) {
        const int buf = t & 1;
        const bool has_next = (t + 1 < nT);

        // 1) issue next tile's global loads into registers (latency hidden under WMMAs)
        if (has_next) {
            const size_t koff = (size_t)(t + 1) * BK;
            #pragma unroll
            for (int p = 0; p < 4; ++p) {
                fa[p] = *(const float4*)(xbase + (size_t)p * 32 * K + koff);
                wb[p] = *(const int32_t*)(wbase + (size_t)p * 32 * K + koff);
            }
        }
        // prefetch two tiles ahead into L2 (global_prefetch_b8)
        if (t + 2 < nT) {
            const size_t koff2 = (size_t)(t + 2) * BK;
            __builtin_prefetch(xbase + koff2, 0, 3);
            __builtin_prefetch(wbase + koff2, 0, 3);
        }

        // 2) fragment loads from current buffer + 8x v_wmma_f32_16x16x32_f16
        const _Float16* Ab = As[buf];
        const _Float16* Bb = Bs[buf];
        v16h a_frag[2], b_frag[4];
        #pragma unroll
        for (int mt = 0; mt < 2; ++mt) {
            const int row = wm * 32 + mt * 16 + lh;
            FragBits fb;
            fb.lo = *(const uint4*)(&Ab[row * LDSS + asel]);
            fb.hi = *(const uint4*)(&Ab[row * LDSS + asel + 16]);
            a_frag[mt] = __builtin_bit_cast(v16h, fb);
        }
        #pragma unroll
        for (int nt = 0; nt < 4; ++nt) {
            const int row = wn * 64 + nt * 16 + lh;   // weight row == B column
            FragBits fb;
            fb.lo = *(const uint4*)(&Bb[row * LDSS + bsel]);
            fb.hi = *(const uint4*)(&Bb[row * LDSS + bsel + 16]);
            b_frag[nt] = __builtin_bit_cast(v16h, fb);
        }
        #pragma unroll
        for (int mt = 0; mt < 2; ++mt) {
            #pragma unroll
            for (int nt = 0; nt < 4; ++nt) {
                acc[mt][nt] = __builtin_amdgcn_wmma_f32_16x16x32_f16(
                    /*neg_a=*/false, a_frag[mt],
                    /*neg_b=*/false, b_frag[nt],
                    /*c_mod=*/(short)0, acc[mt][nt],
                    /*reuse_a=*/false, /*reuse_b=*/false);
            }
        }

        // 3) convert + store next tile into the other buffer (conversions co-execute w/ XDL)
        if (has_next) {
            _Float16* An = As[buf ^ 1];
            _Float16* Bn = Bs[buf ^ 1];
            #pragma unroll
            for (int p = 0; p < 4; ++p) {
                const int row = r0 + p * 32;
                v4h ha;
                ha.x = (_Float16)fa[p].x; ha.y = (_Float16)fa[p].y;
                ha.z = (_Float16)fa[p].z; ha.w = (_Float16)fa[p].w;
                *(v4h*)(&An[row * LDSS + q * 4]) = ha;
                v4h hb;
                hb.x = (_Float16)(int8_t)( wb[p]        & 0xff);
                hb.y = (_Float16)(int8_t)((wb[p] >>  8) & 0xff);
                hb.z = (_Float16)(int8_t)((wb[p] >> 16) & 0xff);
                hb.w = (_Float16)(int8_t)((wb[p] >> 24) & 0xff);
                *(v4h*)(&Bn[row * LDSS + q * 4]) = hb;
            }
            __syncthreads();    // single barrier per K-step (ping-pong makes it safe)
        }
    }

    // ---- epilogue: apply per-output-channel scale post-accumulation ----
    #pragma unroll
    for (int nt = 0; nt < 4; ++nt) {
        const int n = block_n + wn * 64 + nt * 16 + lh;
        const float s = Sc[n];
        #pragma unroll
        for (int mt = 0; mt < 2; ++mt) {
            const int mbase = block_m + wm * 32 + mt * 16 + ((lane < 16) ? 0 : 8);
            #pragma unroll
            for (int r = 0; r < 8; ++r) {
                Out[(size_t)(mbase + r) * N + n] = acc[mt][nt][r] * s;
            }
        }
    }
}

extern "C" void kernel_launch(void* const* d_in, const int* in_sizes, int n_in,
                              void* d_out, int out_size, void* d_ws, size_t ws_size,
                              hipStream_t stream) {
    const float*  X  = (const float*)d_in[0];       // x: [4,2048,4096] fp32
    const int8_t* W  = (const int8_t*)d_in[1];      // weight: [4096,4096] int8
    const float*  Sc = (const float*)d_in[2];       // weight_scaler: [4096] fp32
    float*        O  = (float*)d_out;               // out: [4,2048,4096] fp32

    const int N = in_sizes[2];            // D_OUT = 4096
    const int K = in_sizes[1] / N;        // D_IN  = 4096
    const int M = in_sizes[0] / K;        // B*S   = 8192

    dim3 grid(N / BN, M / BM);            // 32 x 64 workgroups
    int8_linear_wmma_f16<<<grid, dim3(256), 0, stream>>>(X, W, Sc, O, M, N, K);
}
